// FourDirectionalCrossModalScan_70738111365379
// MI455X (gfx1250) — matrix operand order repaired and checked by reading, not compiled, not run on hardware
//
#include <hip/hip_runtime.h>
#include <hip/hip_bf16.h>

// ---------------- constants (match reference) ----------------
#define DIMD   256
#define DI     512              // D_INNER
#define STN    64               // STATE
#define NCAT   (3*DI + 2*STN)   // 1664 = [x | z | dt | B | C]
#define NLAYERS 2
#define BSZ    4
#define SH_    32
#define SW_    64
#define NTOK   (SH_*SW_)        // 2048
#define TTOK   16384            // tokens per bicms pass (both dirs): 128*128 = 256*64
#define EPSV   1e-5f

typedef _Float16 half_t;
typedef __attribute__((ext_vector_type(16))) _Float16 v16h;
typedef __attribute__((ext_vector_type(8)))  _Float16 v8h;
typedef __attribute__((ext_vector_type(8)))  float    v8f;

// ---------------- WMMA f16 GEMM: D = A(MxK) * Bt(NxK, transposed) [+ Cin] ----------------
// One wave computes a 16(M) x 64(N) tile: 4 accumulators share one A fragment per K-step.
// All fragment loads (2x A b128 + 8x B b128) are issued BEFORE the 4 back-to-back WMMAs
// so the loads batch into one clause and the XDL pipe stays busy.
__global__ void gemm_wmma_kernel(const half_t* __restrict__ A,
                                 const half_t* __restrict__ Bt,
                                 const float* Cin, float* D,
                                 int M, int N, int K) {
  int wave = (int)((blockIdx.x * blockDim.x + threadIdx.x) >> 5);
  int lane = threadIdx.x & 31;
  int tilesN = N >> 6;                       // 64-wide N tiles
  int ntiles = (M >> 4) * tilesN;
  if (wave >= ntiles) return;                // wave-uniform exit (EXEC all-1 inside)
  int m0 = (wave / tilesN) << 4;
  int n0 = (wave % tilesN) << 6;

  int g   = lane >> 4;                       // lane half (0/1)
  int l15 = lane & 15;
  int rbase = m0 + (g << 3);

  v8f acc[4];
  if (Cin) {
#pragma unroll
    for (int t = 0; t < 4; ++t) {
      int col = n0 + (t << 4) + l15;
#pragma unroll
      for (int r = 0; r < 8; ++r) acc[t][r] = Cin[(size_t)(rbase + r) * N + col];
    }
  } else {
#pragma unroll
    for (int t = 0; t < 4; ++t)
#pragma unroll
      for (int r = 0; r < 8; ++r) acc[t][r] = 0.f;
  }

  // A: lane holds row m0+l15, K halves {kb+8g..+7} and {kb+16+8g..+7}  (ISA 7.12.2)
  const half_t* Ap = A + (size_t)(m0 + l15) * K + (g << 3);
  // Bt: lane holds col n0+16t+l15, K halves {kb+8g..+7} and {kb+16+8g..+7}
  const half_t* Bp0 = Bt + (size_t)(n0 + l15) * K + (g << 3);
  const size_t bstride = (size_t)16 * K;

  for (int kb = 0; kb < K; kb += 32) {
    // ---- issue ALL fragment loads first ----
    v8h alo = *(const v8h*)(Ap + kb);
    v8h ahi = *(const v8h*)(Ap + kb + 16);
    v16h bf[4];
#pragma unroll
    for (int t = 0; t < 4; ++t) {
      const half_t* bp = Bp0 + (size_t)t * bstride;
      v8h blo = *(const v8h*)(bp + kb);
      v8h bhi = *(const v8h*)(bp + kb + 16);
      bf[t] = __builtin_shufflevector(blo, bhi, 0, 1, 2, 3, 4, 5, 6, 7,
                                      8, 9, 10, 11, 12, 13, 14, 15);
    }
    v16h a = __builtin_shufflevector(alo, ahi, 0, 1, 2, 3, 4, 5, 6, 7,
                                     8, 9, 10, 11, 12, 13, 14, 15);
    if (kb + 32 < K) __builtin_prefetch(Ap + kb + 32, 0, 0);
    // ---- back-to-back WMMAs ----
#pragma unroll
    for (int t = 0; t < 4; ++t)
      acc[t] = __builtin_amdgcn_wmma_f32_16x16x32_f16(false, a, false, bf[t],
                                                      (short)0, acc[t], false, false);
  }
#pragma unroll
  for (int t = 0; t < 4; ++t) {
    int col = n0 + (t << 4) + l15;
#pragma unroll
    for (int r = 0; r < 8; ++r) D[(size_t)(rbase + r) * N + col] = acc[t][r];
  }
}

// ---------------- LayerNorm -> f16 (one token per 256-thread block) ----------------
__global__ void ln_to_f16_kernel(const float* __restrict__ X,
                                 const float* __restrict__ sc,
                                 const float* __restrict__ bi,
                                 half_t* __restrict__ H16) {
  __shared__ float red[DIMD];
  int t = blockIdx.x, d = threadIdx.x;
  float v = X[(size_t)t * DIMD + d];
  red[d] = v; __syncthreads();
  for (int off = 128; off > 0; off >>= 1) { if (d < off) red[d] += red[d + off]; __syncthreads(); }
  float mu = red[0] * (1.f / DIMD);
  __syncthreads();
  float xc = v - mu;
  red[d] = xc * xc; __syncthreads();
  for (int off = 128; off > 0; off >>= 1) { if (d < off) red[d] += red[d + off]; __syncthreads(); }
  float var = red[0] * (1.f / DIMD);
  H16[(size_t)t * DIMD + d] = (half_t)(xc * rsqrtf(var + EPSV) * sc[d] + bi[d]);
}

// ---------------- bias + LayerNorm -> f32 out (final merges) ----------------
__global__ void bias_ln_kernel(const float* __restrict__ X,
                               const float* __restrict__ bias,
                               const float* __restrict__ sc,
                               const float* __restrict__ bi,
                               float* __restrict__ Out) {
  __shared__ float red[DIMD];
  int t = blockIdx.x, d = threadIdx.x;
  float v = X[(size_t)t * DIMD + d] + bias[d];
  red[d] = v; __syncthreads();
  for (int off = 128; off > 0; off >>= 1) { if (d < off) red[d] += red[d + off]; __syncthreads(); }
  float mu = red[0] * (1.f / DIMD);
  __syncthreads();
  float xc = v - mu;
  red[d] = xc * xc; __syncthreads();
  for (int off = 128; off > 0; off >>= 1) { if (d < off) red[d] += red[d + off]; __syncthreads(); }
  float var = red[0] * (1.f / DIMD);
  Out[(size_t)t * DIMD + d] = xc * rsqrtf(var + EPSV) * sc[d] + bi[d];
}

// ---------------- pointwise: xs = silu(xin), dt = softplus(dtlin + bias) ----------------
__global__ void mamba_pointwise_kernel(const float* __restrict__ lin,
                                       const float* __restrict__ dt_bias,
                                       float* __restrict__ xs, float* __restrict__ dtb) {
  size_t idx = (size_t)blockIdx.x * blockDim.x + threadIdx.x;
  if (idx >= (size_t)TTOK * DI) return;
  size_t t = idx / DI; int d = (int)(idx % DI);
  const float* row = lin + t * NCAT;
  float xin = row[d];
  xs[idx] = xin / (1.f + expf(-xin));
  float dl = row[2 * DI + d] + dt_bias[d];
  dtb[idx] = (dl > 20.f) ? dl : log1pf(expf(dl));
}

// ---------------- selective scan: one sequence per block, one channel per thread ----------------
// state h[STN] lives in VGPRs; B_t/C_t staged into LDS via CDNA5 async global->LDS loads.
__global__ void scan_kernel(const float* __restrict__ lin,
                            const float* __restrict__ xs,
                            const float* __restrict__ dtb,
                            const float* __restrict__ A_log,
                            float* __restrict__ y, int L2) {
  __shared__ float BC[2 * STN];          // [0..63] = B_t, [64..127] = C_t
  int s = blockIdx.x, d = threadIdx.x;   // blockDim.x == DI
  float Ad = -expf(A_log[d]);
  float hst[STN];
#pragma unroll
  for (int j = 0; j < STN; ++j) hst[j] = 0.f;
  for (int t = 0; t < L2; ++t) {
    size_t tok = (size_t)s * L2 + t;
    // async global->LDS broadcast of B_t (64) and C_t (64): one b32 per lane d<128
    if (d < 2 * STN) {
      const float* gsrc = lin + tok * NCAT + 3 * DI + d;
      unsigned lds = (unsigned)(size_t)(&BC[0]) + (unsigned)d * 4u;
      asm volatile("global_load_async_to_lds_b32 %0, %1, off"
                   :: "v"(lds), "v"(gsrc) : "memory");
    }
    asm volatile("s_wait_asynccnt 0x0" ::: "memory");
    __syncthreads();
    float dt = dtb[tok * DI + d];
    float xv = xs[tok * DI + d];
    float e  = expf(dt * Ad);
    float dx = dt * xv;
    float acc = 0.f;
#pragma unroll
    for (int j = 0; j < STN; ++j) {
      hst[j] = e * hst[j] + dx * BC[j];
      acc += hst[j] * BC[STN + j];
    }
    y[tok * DI + d] = acc;
    __syncthreads();
  }
}

// ---------------- u = (y + Dskip*xs) * silu(z)  -> f16 ----------------
__global__ void outpre_kernel(const float* __restrict__ y, const float* __restrict__ xs,
                              const float* __restrict__ lin, const float* __restrict__ Dsk,
                              half_t* __restrict__ u16) {
  size_t idx = (size_t)blockIdx.x * blockDim.x + threadIdx.x;
  if (idx >= (size_t)TTOK * DI) return;
  size_t t = idx / DI; int d = (int)(idx % DI);
  float z = lin[t * NCAT + DI + d];
  float sz = z / (1.f + expf(-z));
  u16[idx] = (half_t)((y[idx] + Dsk[d] * xs[idx]) * sz);
}

// ---------------- interleave two streams into sequences (horizontal/vertical) ----------------
__global__ void interleave_kernel(const float* __restrict__ a, const float* __restrict__ b,
                                  float* __restrict__ inter, int vertical) {
  size_t idx = (size_t)blockIdx.x * blockDim.x + threadIdx.x;
  const size_t total = (size_t)TTOK * DIMD;
  if (idx >= total) return;
  int d = (int)(idx & (DIMD - 1));
  size_t tok = idx >> 8;
  int L2 = vertical ? 64 : 128;
  int s = (int)(tok / L2), r = (int)(tok % L2);
  int l = r >> 1, p = r & 1;
  size_t src;
  if (!vertical) src = ((size_t)s * SW_ + l) * DIMD + d;                 // s = b*H + h
  else { int bb = s >> 6, w = s & 63;                                    // s = b*W + w
         src = ((size_t)bb * NTOK + (size_t)l * SW_ + w) * DIMD + d; }
  inter[idx] = p ? b[src] : a[src];
}

__global__ void deinterleave_kernel(const float* __restrict__ merged,
                                    float* __restrict__ sem_out, float* __restrict__ inst_out,
                                    int vertical) {
  size_t idx = (size_t)blockIdx.x * blockDim.x + threadIdx.x;
  const size_t total = (size_t)TTOK * DIMD;
  if (idx >= total) return;
  int d = (int)(idx & (DIMD - 1));
  size_t tok = idx >> 8;
  int L2 = vertical ? 64 : 128;
  int s = (int)(tok / L2), r = (int)(tok % L2);
  int l = r >> 1, p = r & 1;
  size_t dst;
  if (!vertical) dst = ((size_t)s * SW_ + l) * DIMD + d;
  else { int bb = s >> 6, w = s & 63;
         dst = ((size_t)bb * NTOK + (size_t)l * SW_ + w) * DIMD + d; }
  float v = merged[idx];
  if (p == 0) sem_out[dst] = v; else inst_out[dst] = v;
}

__global__ void flip_kernel(const float* __restrict__ in, float* __restrict__ out, int L2) {
  size_t idx = (size_t)blockIdx.x * blockDim.x + threadIdx.x;
  const size_t total = (size_t)TTOK * DIMD;
  if (idx >= total) return;
  int d = (int)(idx & (DIMD - 1));
  size_t tok = idx >> 8;
  int s = (int)(tok / L2), r = (int)(tok % L2);
  out[idx] = in[((size_t)s * L2 + (L2 - 1 - r)) * DIMD + d];
}

__global__ void gate_merge_kernel(const float* __restrict__ glin, const float* __restrict__ gb,
                                  const float* __restrict__ fwd, const float* __restrict__ bwd,
                                  float* __restrict__ merged) {
  size_t idx = (size_t)blockIdx.x * blockDim.x + threadIdx.x;
  if (idx >= (size_t)TTOK * DIMD) return;
  int d = (int)(idx & (DIMD - 1));
  float g = 1.f / (1.f + expf(-(glin[idx] + gb[d])));
  merged[idx] = g * fwd[idx] + (1.f - g) * bwd[idx];
}

// concat two f32 (T x 256) -> f16 (T x 512)
__global__ void concat_f16_kernel(const float* __restrict__ a, const float* __restrict__ b,
                                  half_t* __restrict__ dst, int T) {
  size_t idx = (size_t)blockIdx.x * blockDim.x + threadIdx.x;
  if (idx >= (size_t)T * 2 * DIMD) return;
  size_t t = idx / (2 * DIMD); int d = (int)(idx % (2 * DIMD));
  float v = (d < DIMD) ? a[t * DIMD + d] : b[t * DIMD + (d - DIMD)];
  dst[idx] = (half_t)v;
}

// pack transposed f16 [Wx|Wz|Wdt|WB|WC]^T per (stack,layer): dst[sl][n][k], n in [0,1664), k in [0,256)
__global__ void pack_wcat_t_kernel(const float* __restrict__ Wx, const float* __restrict__ Wz,
                                   const float* __restrict__ Wdt, const float* __restrict__ WB,
                                   const float* __restrict__ WC, half_t* __restrict__ Wcat) {
  size_t idx = (size_t)blockIdx.x * blockDim.x + threadIdx.x;
  const size_t total = (size_t)8 * NCAT * DIMD;
  if (idx >= total) return;
  int k = (int)(idx % DIMD);
  size_t rem = idx / DIMD;
  int n = (int)(rem % NCAT);
  int sl = (int)(rem / NCAT);
  float v;
  if      (n < DI)            v = Wx [((size_t)sl * DIMD + k) * DI  + n];
  else if (n < 2 * DI)        v = Wz [((size_t)sl * DIMD + k) * DI  + (n - DI)];
  else if (n < 3 * DI)        v = Wdt[((size_t)sl * DIMD + k) * DI  + (n - 2 * DI)];
  else if (n < 3 * DI + STN)  v = WB [((size_t)sl * DIMD + k) * STN + (n - 3 * DI)];
  else                        v = WC [((size_t)sl * DIMD + k) * STN + (n - 3 * DI - STN)];
  Wcat[idx] = (half_t)v;
}

// transpose-convert: src (SL, K, N) f32 -> dst (SL, N, K) f16
__global__ void cvt_t_f16_kernel(const float* __restrict__ src, half_t* __restrict__ dst,
                                 int SL, int K, int N) {
  size_t idx = (size_t)blockIdx.x * blockDim.x + threadIdx.x;
  size_t total = (size_t)SL * K * N;
  if (idx >= total) return;
  int k = (int)(idx % K);
  size_t rem = idx / K;
  int n = (int)(rem % N);
  int sl = (int)(rem / N);
  dst[idx] = (half_t)src[((size_t)sl * K + k) * N + n];
}

// ---------------- host orchestration ----------------
namespace {

static inline int gemm_blocks(int M, int N) {
  long waves = (long)(M / 16) * (N / 64);
  return (int)((waves * 32 + 255) / 256);
}
static inline int ew_blocks(size_t n) { return (int)((n + 255) / 256); }

struct Params {
  const float *ln_s, *ln_b, *dt_bias, *A_log, *Dskip;
  const half_t *wcat16, *wout16;
  float *lin, *xs, *dtb, *ybuf;
  half_t *h16, *u16;
};

static void run_stack(float* xb, int si, int L2, const Params& P, hipStream_t stream) {
  const int T = TTOK;
  for (int l = 0; l < NLAYERS; ++l) {
    int sl = si * NLAYERS + l;
    ln_to_f16_kernel<<<T, DIMD, 0, stream>>>(xb, P.ln_s + (size_t)sl * DIMD,
                                             P.ln_b + (size_t)sl * DIMD, P.h16);
    gemm_wmma_kernel<<<gemm_blocks(T, NCAT), 256, 0, stream>>>(
        P.h16, P.wcat16 + (size_t)sl * NCAT * DIMD, nullptr, P.lin, T, NCAT, DIMD);
    mamba_pointwise_kernel<<<ew_blocks((size_t)T * DI), 256, 0, stream>>>(
        P.lin, P.dt_bias + (size_t)sl * DI, P.xs, P.dtb);
    scan_kernel<<<T / L2, DI, 0, stream>>>(P.lin, P.xs, P.dtb,
                                           P.A_log + (size_t)sl * DI, P.ybuf, L2);
    outpre_kernel<<<ew_blocks((size_t)T * DI), 256, 0, stream>>>(
        P.ybuf, P.xs, P.lin, P.Dskip + (size_t)sl * DI, P.u16);
    gemm_wmma_kernel<<<gemm_blocks(T, DIMD), 256, 0, stream>>>(
        P.u16, P.wout16 + (size_t)sl * DIMD * DI, xb, xb, T, DIMD, DI);
  }
}

} // namespace

extern "C" void kernel_launch(void* const* d_in, const int* in_sizes, int n_in,
                              void* d_out, int out_size, void* d_ws, size_t ws_size,
                              hipStream_t stream) {
  (void)in_sizes; (void)n_in; (void)out_size; (void)ws_size;
  const float* stream_sem  = (const float*)d_in[0];
  const float* stream_inst = (const float*)d_in[1];
  const float* Wx     = (const float*)d_in[2];
  const float* Wz     = (const float*)d_in[3];
  const float* Wdt    = (const float*)d_in[4];
  const float* WB     = (const float*)d_in[5];
  const float* WC     = (const float*)d_in[6];
  const float* A_log  = (const float*)d_in[7];
  const float* Dskip  = (const float*)d_in[8];
  const float* dt_bias= (const float*)d_in[9];
  const float* Wout   = (const float*)d_in[10];
  const float* ln_s   = (const float*)d_in[11];
  const float* ln_b   = (const float*)d_in[12];
  const float* gate_W = (const float*)d_in[13];
  const float* gate_b = (const float*)d_in[14];
  const float* merge_W= (const float*)d_in[15];
  const float* merge_b= (const float*)d_in[16];
  const float* mln_s  = (const float*)d_in[17];
  const float* mln_b  = (const float*)d_in[18];
  float* out = (float*)d_out;

  // ---- workspace carve ----
  char* wp = (char*)d_ws;
  auto alloc = [&](size_t bytes) -> void* {
    void* r = (void*)wp; wp += (bytes + 255) & ~(size_t)255; return r;
  };
  const size_t T = TTOK;
  float* inter = (float*)alloc(T * DIMD * 4);
  float* fwdb  = (float*)alloc(T * DIMD * 4);
  float* tmpf  = (float*)alloc(T * DIMD * 4);
  float* bwdb  = (float*)alloc(T * DIMD * 4);
  float* lin   = (float*)alloc(T * NCAT * 4);
  float* xs    = (float*)alloc(T * DI * 4);
  float* dtb   = (float*)alloc(T * DI * 4);
  float* ybuf  = (float*)alloc(T * DI * 4);
  float* glin  = (float*)alloc(T * DIMD * 4);                 // reused for final merge lin
  float* dbuf[4]; // h_sem, h_inst, v_sem, v_inst : each (B*N x 256)
  for (int i = 0; i < 4; ++i) dbuf[i] = (float*)alloc((size_t)BSZ * NTOK * DIMD * 4);
  half_t* h16    = (half_t*)alloc(T * DIMD * 2);
  half_t* u16    = (half_t*)alloc(T * DI * 2);
  half_t* cat16  = (half_t*)alloc(T * 2 * DIMD * 2);
  half_t* catm16 = (half_t*)alloc((size_t)BSZ * NTOK * 2 * DIMD * 2);
  half_t* wcat16 = (half_t*)alloc((size_t)8 * NCAT * DIMD * 2);   // transposed (n,k)
  half_t* wout16 = (half_t*)alloc((size_t)8 * DIMD * DI * 2);     // transposed (n,k)
  half_t* gw16   = (half_t*)alloc((size_t)2 * DIMD * DI * 2);     // transposed
  half_t* mw16   = (half_t*)alloc((size_t)2 * DIMD * DI * 2);     // transposed

  // ---- pack weights to transposed f16 ----
  pack_wcat_t_kernel<<<ew_blocks((size_t)8 * NCAT * DIMD), 256, 0, stream>>>(Wx, Wz, Wdt, WB, WC, wcat16);
  cvt_t_f16_kernel<<<ew_blocks((size_t)8 * DI * DIMD), 256, 0, stream>>>(Wout, wout16, 8, DI, DIMD);
  cvt_t_f16_kernel<<<ew_blocks((size_t)2 * DI * DIMD), 256, 0, stream>>>(gate_W, gw16, 2, DI, DIMD);
  cvt_t_f16_kernel<<<ew_blocks((size_t)2 * DI * DIMD), 256, 0, stream>>>(merge_W, mw16, 2, DI, DIMD);

  Params P;
  P.ln_s = ln_s; P.ln_b = ln_b; P.dt_bias = dt_bias; P.A_log = A_log; P.Dskip = Dskip;
  P.wcat16 = wcat16; P.wout16 = wout16;
  P.lin = lin; P.xs = xs; P.dtb = dtb; P.ybuf = ybuf; P.h16 = h16; P.u16 = u16;

  const size_t tot = T * DIMD;
  // ---- two BiCMS passes: dir 0 = horizontal (stacks 0/1), dir 1 = vertical (stacks 2/3) ----
  for (int dir = 0; dir < 2; ++dir) {
    int L2 = dir ? 64 : 128;
    interleave_kernel<<<ew_blocks(tot), 256, 0, stream>>>(stream_sem, stream_inst, inter, dir);
    // forward stack
    hipMemcpyAsync(fwdb, inter, tot * 4, hipMemcpyDeviceToDevice, stream);
    run_stack(fwdb, 2 * dir + 0, L2, P, stream);
    // backward stack: flip -> stack -> flip
    flip_kernel<<<ew_blocks(tot), 256, 0, stream>>>(inter, tmpf, L2);
    run_stack(tmpf, 2 * dir + 1, L2, P, stream);
    flip_kernel<<<ew_blocks(tot), 256, 0, stream>>>(tmpf, bwdb, L2);
    // gate merge
    concat_f16_kernel<<<ew_blocks(tot * 2), 256, 0, stream>>>(fwdb, bwdb, cat16, (int)T);
    gemm_wmma_kernel<<<gemm_blocks((int)T, DIMD), 256, 0, stream>>>(
        cat16, gw16 + (size_t)dir * DIMD * DI, nullptr, glin, (int)T, DIMD, DI);
    gate_merge_kernel<<<ew_blocks(tot), 256, 0, stream>>>(glin, gate_b + (size_t)dir * DIMD,
                                                          fwdb, bwdb, inter);
    deinterleave_kernel<<<ew_blocks(tot), 256, 0, stream>>>(inter, dbuf[2 * dir], dbuf[2 * dir + 1], dir);
  }

  // ---- final merges: m=0 sem (h_sem|v_sem), m=1 inst (h_inst|v_inst) ----
  const int TM = BSZ * NTOK; // 8192 tokens
  for (int m = 0; m < 2; ++m) {
    concat_f16_kernel<<<ew_blocks((size_t)TM * 2 * DIMD), 256, 0, stream>>>(
        dbuf[m], dbuf[2 + m], catm16, TM);
    gemm_wmma_kernel<<<gemm_blocks(TM, DIMD), 256, 0, stream>>>(
        catm16, mw16 + (size_t)m * DIMD * DI, nullptr, glin, TM, DIMD, DI);
    bias_ln_kernel<<<TM, DIMD, 0, stream>>>(glin, merge_b + (size_t)m * DIMD,
                                            mln_s + (size_t)m * DIMD, mln_b + (size_t)m * DIMD,
                                            out + (size_t)m * TM * DIMD);
  }
}
